// T2A_73727408603250
// MI455X (gfx1250) — matrix operand order repaired and checked by reading, not compiled, not run on hardware
//
// T5-small encoder + 32-step greedy decode for MI455X (gfx1250, wave32, WMMA).
// All GEMMs use v_wmma_f32_16x16x32_f16 (f32 operands converted to f16 in
// registers, f32 accumulation). Each GEMM wave now computes a 16x64 tile with
// 4 accumulators, reusing the converted A fragment across 4 WMMAs per K-step
// (4x less A traffic / cvt work per matrix op). Encoder attention is fused
// (scores->softmax->PV) with a 16x512 LDS row buffer per wave. lm_head (65MB)
// + embed (65MB) both fit in the 192MB L2, so decode replays out of L2.

#include <hip/hip_runtime.h>
#include <math.h>

static constexpr int B   = 8;
static constexpr int S   = 512;
static constexpr int T   = 32;
static constexpr int L   = 6;
static constexpr int H   = 8;
static constexpr int HD  = 64;
static constexpr int D   = 512;
static constexpr int DFF = 2048;
static constexpr int V   = 32128;
static constexpr float NEGF   = -1e9f;
static constexpr float SMOOTH = 0.1f;

typedef __attribute__((ext_vector_type(16))) _Float16 v16h;
typedef __attribute__((ext_vector_type(8)))  float    v8f;

// ---------------------------------------------------------------------------
// WMMA GEMM: C[M,N] = act(A[M,K] @ W[K,N] (+R)).  One 16x64 tile per wave,
// 4 chained v_wmma_f32_16x16x32_f16 per K-step sharing one A fragment.
// Requires N % 64 == 0 (true for all GEMMs here: 512 / 2048 / 32128).
// A-fragment (16-bit A 16x32, same k order as B): lane&15 = M row; lanes 0-15
// cover K 0..15, lanes 16-31 cover K 16..31.
// B-fragment: lane&15 = N col; lanes 0-15 hold K 0..15, lanes 16-31 K 16..31.
// C/D: VGPR r -> M = r (lanes 0-15) / 8+r (lanes 16-31).
// ---------------------------------------------------------------------------
template <bool RELU, bool ADD>
__global__ void k_gemm(const float* __restrict__ A, int lda,
                       const float* __restrict__ W,
                       const float* __restrict__ R,
                       float* __restrict__ C, int ldc,
                       int M, int N, int K)
{
  int wave = (int)((blockIdx.x * blockDim.x + threadIdx.x) >> 5);
  int lane = threadIdx.x & 31;
  int mt = (M + 15) >> 4, nt = N >> 6;          // 16x64 tiles
  if (wave >= mt * nt) return;
  int m0 = (wave / nt) << 4;
  int n0 = (wave % nt) << 6;
  int half = lane >> 4, l16 = lane & 15;

  int am   = m0 + l16;
  bool aok = am < M;
  const float* arow  = A + (size_t)(aok ? am : 0) * lda;
  const float* wbase = W + (size_t)(half * 16) * N + n0 + l16;

  v8f acc[4] = {};
  for (int k0 = 0; k0 < K; k0 += 32) {
    v16h af;
#pragma unroll
    for (int j = 0; j < 8; ++j) {
      af[j]     = (_Float16)(aok ? arow[k0 + half * 16 + j] : 0.0f);
      af[j + 8] = (_Float16)(aok ? arow[k0 + half * 16 + 8 + j] : 0.0f);
    }
    const float* wp = wbase + (size_t)k0 * N;
    if (k0 + 32 < K) __builtin_prefetch(wp + (size_t)32 * N, 0, 0);  // global_prefetch_b8
#pragma unroll
    for (int sN = 0; sN < 4; ++sN) {
      v16h bf;
#pragma unroll
      for (int j = 0; j < 16; ++j) bf[j] = (_Float16)wp[(size_t)j * N + sN * 16];
      acc[sN] = __builtin_amdgcn_wmma_f32_16x16x32_f16(false, af, false, bf,
                                                       (short)0, acc[sN], false, false);
    }
  }
#pragma unroll
  for (int sN = 0; sN < 4; ++sN) {
#pragma unroll
    for (int r = 0; r < 8; ++r) {
      int m = m0 + half * 8 + r;
      if (m < M) {
        size_t o = (size_t)m * ldc + n0 + sN * 16 + l16;
        float vv = acc[sN][r];
        if (ADD)  vv += R[o];
        if (RELU) vv = fmaxf(vv, 0.0f);
        C[o] = vv;
      }
    }
  }
}

// ---------------------------------------------------------------------------
// Fused encoder self-attention. One wave per (b, h, 16-row query tile).
// q/k/v are [B,S,H,HD] row-major (i.e. GEMM outputs [B*S, D]).
// ---------------------------------------------------------------------------
__global__ void k_enc_attn(const float* __restrict__ q, const float* __restrict__ k,
                           const float* __restrict__ v, const float* __restrict__ bias,
                           const int* __restrict__ amask, float* __restrict__ ctx)
{
  __shared__ float sc[16][S];  // 32 KB of the 320 KB WGP LDS
  int bid = blockIdx.x;        // B*H*(S/16) blocks of 32 threads
  int qt = bid % (S / 16);
  int h  = (bid / (S / 16)) % H;
  int b  = bid / ((S / 16) * H);
  int lane = threadIdx.x;
  int half = lane >> 4, l16 = lane & 15;
  int q0 = qt * 16;

  // Q fragments (A layout), HD=64 -> two K=32 fragments.
  const float* qrow = q + ((size_t)(b * S + q0 + l16)) * D + h * HD;
  v16h aq0, aq1;
#pragma unroll
  for (int j = 0; j < 8; ++j) {
    aq0[j]     = (_Float16)qrow[half * 16 + j];
    aq0[j + 8] = (_Float16)qrow[half * 16 + 8 + j];
    aq1[j]     = (_Float16)qrow[32 + half * 16 + j];
    aq1[j + 8] = (_Float16)qrow[32 + half * 16 + 8 + j];
  }
  // scores = Q @ K^T + bias + mask   (T5: no 1/sqrt(d))
  for (int kt = 0; kt < S / 16; ++kt) {
    int kpos = kt * 16 + l16;
    const float* krow = k + ((size_t)(b * S + kpos)) * D + h * HD;
    v16h b0, b1;
#pragma unroll
    for (int j = 0; j < 16; ++j) {
      b0[j] = (_Float16)krow[half * 16 + j];
      b1[j] = (_Float16)krow[32 + half * 16 + j];
    }
    v8f acc = {};
    acc = __builtin_amdgcn_wmma_f32_16x16x32_f16(false, aq0, false, b0, (short)0, acc, false, false);
    acc = __builtin_amdgcn_wmma_f32_16x16x32_f16(false, aq1, false, b1, (short)0, acc, false, false);
    float mk = amask[b * S + kpos] ? 0.0f : NEGF;
#pragma unroll
    for (int r = 0; r < 8; ++r) {
      int qr = half * 8 + r;
      sc[qr][kpos] = acc[r] + bias[((size_t)h * S + q0 + qr) * S + kpos] + mk;
    }
  }
  __syncthreads();
  // row softmax (lanes 0..15, one query row each)
  if (lane < 16) {
    float m = -1e30f;
    for (int s = 0; s < S; ++s) m = fmaxf(m, sc[lane][s]);
    float sum = 0.0f;
    for (int s = 0; s < S; ++s) { float e = __expf(sc[lane][s] - m); sc[lane][s] = e; sum += e; }
    float inv = 1.0f / sum;
    for (int s = 0; s < S; ++s) sc[lane][s] *= inv;
  }
  __syncthreads();
  // ctx = P[16,S] @ V[S,64]
  v8f co[4] = {};
  for (int kb = 0; kb < S; kb += 32) {
    v16h ap;
#pragma unroll
    for (int j = 0; j < 8; ++j) {
      ap[j]     = (_Float16)sc[l16][kb + half * 16 + j];
      ap[j + 8] = (_Float16)sc[l16][kb + half * 16 + 8 + j];
    }
#pragma unroll
    for (int ntl = 0; ntl < 4; ++ntl) {
      v16h bv;
#pragma unroll
      for (int j = 0; j < 16; ++j)
        bv[j] = (_Float16)v[((size_t)(b * S + kb + half * 16 + j)) * D + h * HD + ntl * 16 + l16];
      co[ntl] = __builtin_amdgcn_wmma_f32_16x16x32_f16(false, ap, false, bv, (short)0,
                                                       co[ntl], false, false);
    }
  }
#pragma unroll
  for (int ntl = 0; ntl < 4; ++ntl)
#pragma unroll
    for (int r = 0; r < 8; ++r)
      ctx[((size_t)(b * S + q0 + half * 8 + r)) * D + h * HD + ntl * 16 + l16] = co[ntl][r];
}

// --------------------------- small helper kernels ---------------------------

__global__ void k_t5_ln(const float* __restrict__ x, const float* __restrict__ w,
                        float* __restrict__ y)
{
  int row = blockIdx.x, tid = threadIdx.x;  // 256 threads / row of D=512
  const float* xr = x + (size_t)row * D;
  float ss = 0.0f;
  for (int d = tid; d < D; d += 256) { float vv = xr[d]; ss += vv * vv; }
  for (int off = 16; off; off >>= 1) ss += __shfl_xor(ss, off);
  __shared__ float red[8];
  if ((tid & 31) == 0) red[tid >> 5] = ss;
  __syncthreads();
  if (tid == 0) {
    float s = 0.0f;
    for (int i = 0; i < 8; ++i) s += red[i];
    red[0] = rsqrtf(s / (float)D + 1e-6f);
  }
  __syncthreads();
  float r = red[0];
  float* yr = y + (size_t)row * D;
  for (int d = tid; d < D; d += 256) yr[d] = xr[d] * w[d] * r;
}

__global__ void k_embed_lookup(const int* __restrict__ ids, const float* __restrict__ emb,
                               float* __restrict__ x)
{
  int idx = blockIdx.x * blockDim.x + threadIdx.x;
  if (idx >= B * S * D) return;
  int tok = idx / D, d = idx - tok * D;
  x[idx] = emb[(size_t)ids[tok] * D + d];
}

__global__ void k_enc_bias(const float* __restrict__ relpos, float* __restrict__ bias)
{
  int idx = blockIdx.x * blockDim.x + threadIdx.x;
  if (idx >= S * S) return;
  int qq = idx / S, kk = idx % S;
  int n = qq - kk;                      // n = -(k - q)
  int ret = (n < 0) ? 16 : 0;           // bidirectional: nb = NB/2 = 16
  int na = n < 0 ? -n : n;
  int bucket;
  if (na < 8) bucket = ret + na;        // max_exact = 8
  else {
    int large = 8 + (int)(logf((float)na / 8.0f) / logf(16.0f) * 8.0f);
    bucket = ret + (large < 15 ? large : 15);
  }
  for (int h = 0; h < H; ++h)
    bias[((size_t)h * S + qq) * S + kk] = relpos[bucket * H + h];
}

__global__ void k_dec_bias(const float* __restrict__ relpos, float* __restrict__ bias)
{
  int idx = blockIdx.x * blockDim.x + threadIdx.x;
  if (idx >= T * T) return;
  int qq = idx / T, kk = idx % T;
  int n = qq - kk;                      // unidirectional: n = max(-(k-q), 0)
  if (n < 0) n = 0;
  int bucket;
  if (n < 16) bucket = n;               // max_exact = 16 (nb = 32)
  else {
    int large = 16 + (int)(logf((float)n / 16.0f) / logf(8.0f) * 16.0f);
    bucket = large < 31 ? large : 31;
  }
  for (int h = 0; h < H; ++h)
    bias[((size_t)h * T + qq) * T + kk] = relpos[bucket * H + h];
}

__global__ void k_init_emb(const float* __restrict__ embed, float* __restrict__ emb)
{
  int idx = blockIdx.x * blockDim.x + threadIdx.x;
  if (idx >= B * D) return;
  emb[idx] = embed[idx % D];            // embed[PAD=0] broadcast
}

__global__ void k_kv_write(const float* __restrict__ kb, const float* __restrict__ vb,
                           float* __restrict__ kc, float* __restrict__ vc, int t)
{
  int idx = blockIdx.x * blockDim.x + threadIdx.x;   // over B*D, layout (b,h,hd)
  if (idx >= B * D) return;
  int b = idx / D, h = (idx % D) / HD, e = idx % HD;
  size_t dst = (((size_t)b * H + h) * T + t) * HD + e;
  kc[dst] = kb[idx];
  vc[dst] = vb[idx];
}

__global__ void k_dec_self_attn(const float* __restrict__ q, const float* __restrict__ kc,
                                const float* __restrict__ vc, const float* __restrict__ bias,
                                int t, float* __restrict__ ctx)
{
  int b = blockIdx.x / H, h = blockIdx.x % H;
  int lane = threadIdx.x;               // 32 lanes == T cache slots
  __shared__ float a_sh[T];
  const float* qv = q + (size_t)b * D + h * HD;
  const float* kr = kc + (((size_t)b * H + h) * T + lane) * HD;
  float s = 0.0f;
  for (int e = 0; e < HD; ++e) s += qv[e] * kr[e];
  s += bias[((size_t)h * T + t) * T + lane];
  if (lane > t) s = NEGF;
  float m = s;
  for (int off = 16; off; off >>= 1) m = fmaxf(m, __shfl_xor(m, off));
  float e = __expf(s - m);
  float sum = e;
  for (int off = 16; off; off >>= 1) sum += __shfl_xor(sum, off);
  a_sh[lane] = e / sum;
  __syncthreads();
  for (int hh = lane; hh < HD; hh += 32) {
    float acc = 0.0f;
    for (int tau = 0; tau <= t; ++tau)
      acc += a_sh[tau] * vc[(((size_t)b * H + h) * T + tau) * HD + hh];
    ctx[(size_t)b * D + h * HD + hh] = acc;
  }
}

__global__ void k_dec_cross_score(const float* __restrict__ q, const float* __restrict__ ck,
                                  const int* __restrict__ amask, float* __restrict__ sc)
{
  int idx = blockIdx.x * blockDim.x + threadIdx.x;   // over B*H*S
  if (idx >= B * H * S) return;
  int s = idx % S, h = (idx / S) % H, b = idx / (S * H);
  const float* qv = q + (size_t)b * D + h * HD;
  const float* kr = ck + ((size_t)(b * S + s)) * D + h * HD;
  float acc = 0.0f;
  for (int e = 0; e < HD; ++e) acc += qv[e] * kr[e];
  sc[idx] = acc + (amask[b * S + s] ? 0.0f : NEGF);
}

__global__ void k_dec_cross_ctx(const float* __restrict__ sc, const float* __restrict__ cv,
                                float* __restrict__ ctx)
{
  int b = blockIdx.x / H, h = blockIdx.x % H;
  int lane = threadIdx.x;               // one wave per (b,h)
  const float* row = sc + ((size_t)b * H + h) * S;
  float m = -1e30f;
  for (int s = lane; s < S; s += 32) m = fmaxf(m, row[s]);
  for (int off = 16; off; off >>= 1) m = fmaxf(m, __shfl_xor(m, off));
  float sum = 0.0f;
  for (int s = lane; s < S; s += 32) sum += __expf(row[s] - m);
  for (int off = 16; off; off >>= 1) sum += __shfl_xor(sum, off);
  float inv = 1.0f / sum;
  for (int hh = lane; hh < HD; hh += 32) {
    float acc = 0.0f;
    for (int s = 0; s < S; ++s)
      acc += __expf(row[s] - m) * inv * cv[((size_t)(b * S + s)) * D + h * HD + hh];
    ctx[(size_t)b * D + h * HD + hh] = acc;
  }
}

__global__ void k_vocab_softmax(const float* __restrict__ logits, const float* __restrict__ tmask,
                                float* __restrict__ probs_out, float* __restrict__ smoothed,
                                float* __restrict__ mask_out, int t)
{
  int b = blockIdx.x, tid = threadIdx.x;  // 256 threads / batch row
  const float* lg = logits + (size_t)b * V;
  const float* tm = tmask + (size_t)b * V;
  __shared__ float rm[256];
  __shared__ int   ri[256];
  float m = -1e30f; int mi = V;
  for (int v = tid; v < V; v += 256) {
    float val = lg[v] + tm[v];
    if (val > m) { m = val; mi = v; }
  }
  rm[tid] = m; ri[tid] = mi;
  __syncthreads();
  for (int off = 128; off; off >>= 1) {
    if (tid < off) {
      if (rm[tid + off] > rm[tid] ||
          (rm[tid + off] == rm[tid] && ri[tid + off] < ri[tid])) {
        rm[tid] = rm[tid + off]; ri[tid] = ri[tid + off];
      }
    }
    __syncthreads();
  }
  float gmax = rm[0];
  int amax = ri[0];
  __syncthreads();
  float sum = 0.0f;
  for (int v = tid; v < V; v += 256) sum += __expf(lg[v] + tm[v] - gmax);
  rm[tid] = sum;
  __syncthreads();
  for (int off = 128; off; off >>= 1) {
    if (tid < off) rm[tid] += rm[tid + off];
    __syncthreads();
  }
  float inv = 1.0f / rm[0];
  float* po = probs_out + ((size_t)b * T + t) * V;
  float* sm = smoothed + (size_t)b * V;
  for (int v = tid; v < V; v += 256) {
    float p = __expf(lg[v] + tm[v] - gmax) * inv;
    po[v] = p;
    sm[v] = p * (1.0f - SMOOTH) + SMOOTH / (float)V;
  }
  if (tid == 0) mask_out[b * T + t] = (amax == 0) ? 1.0f : 0.0f;  // pred == PAD
}

__global__ void k_emb_hist(const float* __restrict__ emb, float* __restrict__ out, int t)
{
  int idx = blockIdx.x * blockDim.x + threadIdx.x;
  if (idx >= B * D) return;
  int b = idx / D, d = idx % D;
  out[((size_t)b * T + t) * D + d] = emb[idx];
}

// ------------------------------- host driver -------------------------------

static void gemm(hipStream_t st, const float* A, int lda, const float* W,
                 const float* R, float* C, int ldc, int M, int N, int K,
                 bool relu, bool add)
{
  int mt = (M + 15) >> 4, nt = N >> 6;   // 16x64 tiles per wave (N % 64 == 0)
  long waves = (long)mt * nt;
  int blocks = (int)((waves + 7) / 8);
  dim3 g(blocks), blk(256);
  if (relu)     k_gemm<true, false><<<g, blk, 0, st>>>(A, lda, W, R, C, ldc, M, N, K);
  else if (add) k_gemm<false, true><<<g, blk, 0, st>>>(A, lda, W, R, C, ldc, M, N, K);
  else          k_gemm<false, false><<<g, blk, 0, st>>>(A, lda, W, R, C, ldc, M, N, K);
}

extern "C" void kernel_launch(void* const* d_in, const int* in_sizes, int n_in,
                              void* d_out, int out_size, void* d_ws, size_t ws_size,
                              hipStream_t stream)
{
  (void)in_sizes; (void)n_in; (void)out_size; (void)ws_size;
  const int*   input_ids    = (const int*)d_in[0];
  const int*   amask        = (const int*)d_in[1];
  const float* token_masks  = (const float*)d_in[3];
  const float* embed        = (const float*)d_in[4];
  const float* lm_head      = (const float*)d_in[5];
  const float* enc_attn_w   = (const float*)d_in[6];
  const float* enc_ln       = (const float*)d_in[7];
  const float* enc_ffn_wi   = (const float*)d_in[8];
  const float* enc_ffn_wo   = (const float*)d_in[9];
  const float* enc_final_ln = (const float*)d_in[10];
  const float* enc_relpos   = (const float*)d_in[11];
  const float* dec_self_w   = (const float*)d_in[12];
  const float* dec_cross_w  = (const float*)d_in[13];
  const float* dec_ln       = (const float*)d_in[14];
  const float* dec_ffn_wi   = (const float*)d_in[15];
  const float* dec_ffn_wo   = (const float*)d_in[16];
  const float* dec_final_ln = (const float*)d_in[17];
  const float* dec_relpos   = (const float*)d_in[18];

  float* out_probs = (float*)d_out;                       // [B,T,V]
  float* out_emb   = out_probs + (size_t)B * T * V;       // [B,T,D]
  float* out_mask  = out_emb + (size_t)B * T * D;         // [B,T] as float

  // workspace carve-out (~205 MB), 256B-aligned slices
  char* wp = (char*)d_ws;
  auto alloc = [&](size_t elems) {
    float* r = (float*)wp;
    wp += ((elems * sizeof(float)) + 255) & ~(size_t)255;
    return r;
  };
  const size_t BSD = (size_t)B * S * D;
  float* xe    = alloc(BSD);               // encoder residual stream
  float* hb    = alloc(BSD);               // LN output / enc_h
  float* qb    = alloc(BSD);
  float* kb    = alloc(BSD);
  float* vb    = alloc(BSD);
  float* cb    = alloc(BSD);               // attention context
  float* fb    = alloc((size_t)B * S * DFF);
  float* ebias = alloc((size_t)H * S * S);
  float* ck    = alloc((size_t)L * BSD);   // cross K, [L][B,S,H,HD]
  float* cv    = alloc((size_t)L * BSD);   // cross V
  float* dbias = alloc((size_t)H * T * T);
  float* kcache = alloc((size_t)L * B * H * T * HD);
  float* vcache = alloc((size_t)L * B * H * T * HD);
  float* emb   = alloc((size_t)B * D);     // decoder residual stream / next emb
  float* dh    = alloc((size_t)B * D);
  float* dq    = alloc((size_t)B * D);
  float* dk    = alloc((size_t)B * D);
  float* dv    = alloc((size_t)B * D);
  float* dctx  = alloc((size_t)B * D);
  float* dffb  = alloc((size_t)B * DFF);
  float* logits = alloc((size_t)B * V);
  float* smooth = alloc((size_t)B * V);
  float* scx   = alloc((size_t)B * H * S);

  // ------------------------------ encoder ------------------------------
  k_embed_lookup<<<(B * S * D + 255) / 256, 256, 0, stream>>>(input_ids, embed, xe);
  k_enc_bias<<<(S * S + 255) / 256, 256, 0, stream>>>(enc_relpos, ebias);
  for (int l = 0; l < L; ++l) {
    const float* Wq = enc_attn_w + (size_t)(l * 4 + 0) * D * D;
    const float* Wk = enc_attn_w + (size_t)(l * 4 + 1) * D * D;
    const float* Wv = enc_attn_w + (size_t)(l * 4 + 2) * D * D;
    const float* Wo = enc_attn_w + (size_t)(l * 4 + 3) * D * D;
    k_t5_ln<<<B * S, 256, 0, stream>>>(xe, enc_ln + (size_t)(l * 2 + 0) * D, hb);
    gemm(stream, hb, D, Wq, nullptr, qb, D, B * S, D, D, false, false);
    gemm(stream, hb, D, Wk, nullptr, kb, D, B * S, D, D, false, false);
    gemm(stream, hb, D, Wv, nullptr, vb, D, B * S, D, D, false, false);
    k_enc_attn<<<B * H * (S / 16), 32, 0, stream>>>(qb, kb, vb, ebias, amask, cb);
    gemm(stream, cb, D, Wo, xe, xe, D, B * S, D, D, false, true);        // residual
    k_t5_ln<<<B * S, 256, 0, stream>>>(xe, enc_ln + (size_t)(l * 2 + 1) * D, hb);
    gemm(stream, hb, D, enc_ffn_wi + (size_t)l * D * DFF, nullptr, fb, DFF,
         B * S, DFF, D, true, false);                                    // ReLU
    gemm(stream, fb, DFF, enc_ffn_wo + (size_t)l * DFF * D, xe, xe, D,
         B * S, D, DFF, false, true);                                    // residual
  }
  k_t5_ln<<<B * S, 256, 0, stream>>>(xe, enc_final_ln, hb);              // hb = enc_h

  // cross-attention K/V, once per layer
  for (int l = 0; l < L; ++l) {
    gemm(stream, hb, D, dec_cross_w + (size_t)(l * 4 + 1) * D * D, nullptr,
         ck + (size_t)l * BSD, D, B * S, D, D, false, false);
    gemm(stream, hb, D, dec_cross_w + (size_t)(l * 4 + 2) * D * D, nullptr,
         cv + (size_t)l * BSD, D, B * S, D, D, false, false);
  }

  // ------------------------------ decoder ------------------------------
  k_dec_bias<<<(T * T + 255) / 256, 256, 0, stream>>>(dec_relpos, dbias);
  k_init_emb<<<(B * D + 255) / 256, 256, 0, stream>>>(embed, emb);
  const size_t KVL = (size_t)B * H * T * HD;
  for (int t = 0; t < T; ++t) {
    for (int l = 0; l < L; ++l) {
      const float* Sq = dec_self_w + (size_t)(l * 4 + 0) * D * D;
      const float* Sk = dec_self_w + (size_t)(l * 4 + 1) * D * D;
      const float* Sv = dec_self_w + (size_t)(l * 4 + 2) * D * D;
      const float* So = dec_self_w + (size_t)(l * 4 + 3) * D * D;
      const float* Cq = dec_cross_w + (size_t)(l * 4 + 0) * D * D;
      const float* Co = dec_cross_w + (size_t)(l * 4 + 3) * D * D;
      float* kc_l = kcache + (size_t)l * KVL;
      float* vc_l = vcache + (size_t)l * KVL;
      // self-attention with KV cache
      k_t5_ln<<<B, 256, 0, stream>>>(emb, dec_ln + (size_t)(l * 3 + 0) * D, dh);
      gemm(stream, dh, D, Sq, nullptr, dq, D, B, D, D, false, false);
      gemm(stream, dh, D, Sk, nullptr, dk, D, B, D, D, false, false);
      gemm(stream, dh, D, Sv, nullptr, dv, D, B, D, D, false, false);
      k_kv_write<<<(B * D + 255) / 256, 256, 0, stream>>>(dk, dv, kc_l, vc_l, t);
      k_dec_self_attn<<<B * H, 32, 0, stream>>>(dq, kc_l, vc_l, dbias, t, dctx);
      gemm(stream, dctx, D, So, emb, emb, D, B, D, D, false, true);
      // cross-attention
      k_t5_ln<<<B, 256, 0, stream>>>(emb, dec_ln + (size_t)(l * 3 + 1) * D, dh);
      gemm(stream, dh, D, Cq, nullptr, dq, D, B, D, D, false, false);
      k_dec_cross_score<<<(B * H * S + 255) / 256, 256, 0, stream>>>(
          dq, ck + (size_t)l * BSD, amask, scx);
      k_dec_cross_ctx<<<B * H, 32, 0, stream>>>(scx, cv + (size_t)l * BSD, dctx);
      gemm(stream, dctx, D, Co, emb, emb, D, B, D, D, false, true);
      // FFN
      k_t5_ln<<<B, 256, 0, stream>>>(emb, dec_ln + (size_t)(l * 3 + 2) * D, dh);
      gemm(stream, dh, D, dec_ffn_wi + (size_t)l * D * DFF, nullptr, dffb, DFF,
           B, DFF, D, true, false);
      gemm(stream, dffb, DFF, dec_ffn_wo + (size_t)l * DFF * D, emb, emb, D,
           B, D, DFF, false, true);
    }
    k_t5_ln<<<B, 256, 0, stream>>>(emb, dec_final_ln, dh);
    gemm(stream, dh, D, lm_head, nullptr, logits, V, B, V, D, false, false);
    k_vocab_softmax<<<B, 256, 0, stream>>>(logits, token_masks, out_probs, smooth,
                                           out_mask, t);
    // new_emb = smoothed @ embed  (K = V = 32128, streams embed out of L2)
    gemm(stream, smooth, V, embed, nullptr, emb, D, B, D, V, false, false);
    k_emb_hist<<<(B * D + 255) / 256, 256, 0, stream>>>(emb, out_emb, t);
  }
}
